// SetAbstraction_82171314307281
// MI455X (gfx1250) — matrix operand order repaired and checked by reading, not compiled, not run on hardware
//
#include <hip/hip_runtime.h>

// ---------------- problem constants (match reference) ----------------
#define BATCH 16
#define NPT   4096      // N points per cloud
#define SSAMP 1024      // S sampled centers per cloud
#define CINF  64        // input feature channels
#define KNBR  64        // neighbors per center
#define HIDD  128       // hidden width
#define KP1   96        // padded inner dim for GEMM1 (67 -> 96, zero pad)
#define RAD2  0.04f     // radius^2 = 0.2^2
#define CPB   16        // centers processed per SA block

typedef __bf16 bf16;
typedef __attribute__((ext_vector_type(16))) __bf16 v16bf;
typedef __attribute__((ext_vector_type(8)))  __bf16 v8bf;
typedef __attribute__((ext_vector_type(8)))  float  v8f;

// ---- WMMA fragment loaders (CDNA5 16-bit layouts, CDNA5 ISA) ----
// A 16x32 (MxK): lane L(0-15) row M=L holds K[kb+0..7] then K[kb+16..23];
//                lane L+16 same row holds K[kb+8..15] then K[kb+24..31].
__device__ __forceinline__ v16bf frag_a(const bf16* base, int stride, int mrow,
                                        int kb, int lane) {
  const bf16* p = base + (mrow + (lane & 15)) * stride + kb + ((lane & 16) ? 8 : 0);
  v8bf lo = *(const v8bf*)p;          // 16B aligned -> ds_load_b128
  v8bf hi = *(const v8bf*)(p + 16);
  v16bf r;
#pragma unroll
  for (int i = 0; i < 8; ++i) { r[i] = lo[i]; r[i + 8] = hi[i]; }
  return r;
}
// B 32x16 (KxN), W stored transposed (rows = N columns, contiguous in K):
// lane L(0-15) col N=L holds K[kb+0..7] then K[kb+16..23]; lane L+16 holds K[kb+8..15] then K[kb+24..31].
__device__ __forceinline__ v16bf frag_b(const bf16* baseT, int stride, int ncol,
                                        int kb, int lane) {
  const bf16* p = baseT + (ncol + (lane & 15)) * stride + kb + ((lane & 16) ? 8 : 0);
  v8bf lo = *(const v8bf*)p;
  v8bf hi = *(const v8bf*)(p + 16);
  v16bf r;
#pragma unroll
  for (int i = 0; i < 8; ++i) { r[i] = lo[i]; r[i + 8] = hi[i]; }
  return r;
}

// =====================================================================
// Kernel 1: farthest point sampling, one 1024-thread block per cloud.
// Writes: fps index list (workspace), centers (d_out pos section),
//         batch ids (d_out batch section).
// =====================================================================
__global__ __launch_bounds__(1024) void fps_kernel(const float* __restrict__ pos,
                                                   int* __restrict__ idx_ws,
                                                   float* __restrict__ pos_out,
                                                   int* __restrict__ batch_out) {
  const int b   = blockIdx.x;
  const int tid = threadIdx.x;
  __shared__ float ppx[NPT], ppy[NPT], ppz[NPT];
  __shared__ float rv[1024];
  __shared__ int   ri[1024];
  __shared__ float lpos[3];
  __shared__ int   lidx;

  const float* pcl = pos + (size_t)b * NPT * 3;
  for (int i = tid; i < NPT; i += 1024) {
    ppx[i] = pcl[i * 3 + 0];
    ppy[i] = pcl[i * 3 + 1];
    ppz[i] = pcl[i * 3 + 2];
  }
  if (tid == 0) lidx = 0;
  __syncthreads();
  if (tid == 0) { lpos[0] = ppx[0]; lpos[1] = ppy[0]; lpos[2] = ppz[0]; }
  __syncthreads();

  float mind[4] = {__builtin_inff(), __builtin_inff(), __builtin_inff(), __builtin_inff()};

  for (int s = 0; s < SSAMP; ++s) {
    const int   last = lidx;
    const float lx = lpos[0], ly = lpos[1], lz = lpos[2];
    if (tid == 0) {
      idx_ws[b * SSAMP + s] = last;
      size_t o = ((size_t)b * SSAMP + s) * 3;
      pos_out[o + 0] = lx; pos_out[o + 1] = ly; pos_out[o + 2] = lz;
      batch_out[b * SSAMP + s] = b;
    }
    float bv = -1.0f; int bi = 0;
#pragma unroll
    for (int j = 0; j < 4; ++j) {
      int i = tid + j * 1024;                 // stride-1 per wave: conflict-free LDS
      float dx = ppx[i] - lx, dy = ppy[i] - ly, dz = ppz[i] - lz;
      float d = dx * dx + dy * dy + dz * dz;
      mind[j] = fminf(mind[j], d);
      if (mind[j] > bv) { bv = mind[j]; bi = i; }
    }
    rv[tid] = bv; ri[tid] = bi;
    __syncthreads();
    for (int stp = 512; stp > 0; stp >>= 1) {
      if (tid < stp) {
        float v2 = rv[tid + stp]; int i2 = ri[tid + stp];
        if (v2 > rv[tid] || (v2 == rv[tid] && i2 < ri[tid])) { rv[tid] = v2; ri[tid] = i2; }
      }
      __syncthreads();
    }
    if (tid == 0) {
      int nx = ri[0];
      lidx = nx;
      lpos[0] = ppx[nx]; lpos[1] = ppy[nx]; lpos[2] = ppz[nx];
    }
    __syncthreads();
  }
}

// =====================================================================
// Kernel 2: fused ball-query + 2-layer MLP (bf16 WMMA) + max-pool.
// 256 threads = 8 waves; wave w owns N-tile w (16 cols of 128).
// =====================================================================
__global__ __launch_bounds__(256) void sa_kernel(const float* __restrict__ x,
                                                 const float* __restrict__ pos,
                                                 const float* __restrict__ W1,
                                                 const float* __restrict__ b1,
                                                 const float* __restrict__ W2,
                                                 const float* __restrict__ b2,
                                                 const int* __restrict__ fpsidx,
                                                 float* __restrict__ out) {
  const int tid  = threadIdx.x;
  const int lane = tid & 31;
  const int nt   = tid >> 5;          // wave id == N tile (0..7)

  __shared__ __align__(16) bf16 W1T[HIDD][KP1];    // W1 transposed, K-padded
  __shared__ __align__(16) bf16 W2T[HIDD][HIDD];   // W2 transposed
  __shared__ __align__(16) bf16 In[KNBR][KP1];     // [neighbor][feat||relpos||pad]
  __shared__ __align__(16) bf16 H1[KNBR][HIDD];    // hidden activations
  __shared__ float b1s[HIDD], b2s[HIDD];
  __shared__ int   nbr_s[KNBR];
  __shared__ int   scan_s[256];

  // ---- stage weights/bias once per block ----
  for (int i = tid; i < HIDD * KP1; i += 256) {
    int n = i / KP1, k = i % KP1;
    W1T[n][k] = (k < CINF + 3) ? (bf16)W1[k * HIDD + n] : (bf16)0.0f;
  }
  for (int i = tid; i < HIDD * HIDD; i += 256) {
    int n = i / HIDD, k = i % HIDD;
    W2T[n][k] = (bf16)W2[k * HIDD + n];
  }
  for (int i = tid; i < HIDD; i += 256) { b1s[i] = b1[i]; b2s[i] = b2[i]; }
  __syncthreads();

  for (int ci = 0; ci < CPB; ++ci) {
    const int c = blockIdx.x * CPB + ci;     // global center id = b*S + s
    const int b = c / SSAMP;
    const float* pcl = pos + (size_t)b * NPT * 3;
    const int cidx = fpsidx[c];              // uniform -> scalar load
    const float cx = pcl[cidx * 3 + 0];
    const float cy = pcl[cidx * 3 + 1];
    const float cz = pcl[cidx * 3 + 2];

    // ---- deterministic ball query: first-K in index order within radius ----
    unsigned mask = 0u;
    const int basei = tid * 16;
#pragma unroll
    for (int j = 0; j < 16; ++j) {
      int i = basei + j;
      float dx = pcl[i * 3 + 0] - cx;
      float dy = pcl[i * 3 + 1] - cy;
      float dz = pcl[i * 3 + 2] - cz;
      if (dx * dx + dy * dy + dz * dz <= RAD2) mask |= (1u << j);
    }
    const int mycnt = __popc(mask);
    scan_s[tid] = mycnt;
    __syncthreads();
    for (int off = 1; off < 256; off <<= 1) {     // inclusive Hillis-Steele scan
      int t = (tid >= off) ? scan_s[tid - off] : 0;
      __syncthreads();
      scan_s[tid] += t;
      __syncthreads();
    }
    int excl = scan_s[tid] - mycnt;
    unsigned m = mask;
    while (m) {
      int j = __ffs(m) - 1;
      m &= m - 1;
      if (excl < KNBR) nbr_s[excl] = basei + j;
      ++excl;
    }
    int cnt = scan_s[255];
    if (cnt > KNBR) cnt = KNBR;
    __syncthreads();

    // ---- stage input tile [K, CIN||rel||0] as bf16 ----
    for (int i = tid; i < KNBR * KP1; i += 256) {
      int r = i / KP1, col = i % KP1;
      bf16 v = (bf16)0.0f;
      if (r < cnt) {
        int p = nbr_s[r];
        if (col < CINF) {
          v = (bf16)x[((size_t)b * NPT + p) * CINF + col];
        } else if (col < CINF + 3) {
          float cc = (col == CINF) ? cx : (col == CINF + 1) ? cy : cz;
          v = (bf16)(pcl[p * 3 + (col - CINF)] - cc);
        }
      }
      In[r][col] = v;
    }
    __syncthreads();

    const int ncol  = nt * 16;
    const int n     = ncol + (lane & 15);
    const int rbase = (lane & 16) ? 8 : 0;

    // ---- GEMM1: H1 = relu(In @ W1 + b1), K padded to 96 ----
    v8f acc[4];
#pragma unroll
    for (int mt = 0; mt < 4; ++mt) acc[mt] = (v8f){0.f,0.f,0.f,0.f,0.f,0.f,0.f,0.f};
#pragma unroll
    for (int kb = 0; kb < KP1; kb += 32) {
      v16bf bf = frag_b(&W1T[0][0], KP1, ncol, kb, lane);
#pragma unroll
      for (int mt = 0; mt < 4; ++mt) {
        v16bf af = frag_a(&In[0][0], KP1, mt * 16, kb, lane);
        acc[mt] = __builtin_amdgcn_wmma_f32_16x16x32_bf16(
            false, af, false, bf, (short)0, acc[mt], false, false);
      }
    }
    {
      float bias = b1s[n];
#pragma unroll
      for (int mt = 0; mt < 4; ++mt) {
#pragma unroll
        for (int e = 0; e < 8; ++e) {
          int row = mt * 16 + rbase + e;
          float v = fmaxf(acc[mt][e] + bias, 0.0f);
          H1[row][n] = (bf16)v;
        }
      }
    }
    __syncthreads();

    // ---- GEMM2: H2 = relu(H1 @ W2 + b2); masked max over valid rows ----
#pragma unroll
    for (int mt = 0; mt < 4; ++mt) acc[mt] = (v8f){0.f,0.f,0.f,0.f,0.f,0.f,0.f,0.f};
#pragma unroll
    for (int kb = 0; kb < HIDD; kb += 32) {
      v16bf bf = frag_b(&W2T[0][0], HIDD, ncol, kb, lane);
#pragma unroll
      for (int mt = 0; mt < 4; ++mt) {
        v16bf af = frag_a(&H1[0][0], HIDD, mt * 16, kb, lane);
        acc[mt] = __builtin_amdgcn_wmma_f32_16x16x32_bf16(
            false, af, false, bf, (short)0, acc[mt], false, false);
      }
    }
    {
      float bias = b2s[n];
      float mx = -__builtin_inff();
#pragma unroll
      for (int mt = 0; mt < 4; ++mt) {
#pragma unroll
        for (int e = 0; e < 8; ++e) {
          int row = mt * 16 + rbase + e;
          if (row < cnt) {
            float v = fmaxf(acc[mt][e] + bias, 0.0f);
            mx = fmaxf(mx, v);
          }
        }
      }
      mx = fmaxf(mx, __shfl_xor(mx, 16, 32));   // combine the two half-column sets
      if (lane < 16) out[(size_t)c * HIDD + n] = mx;
    }
    __syncthreads();   // protect LDS reuse across center iterations
  }
}

// =====================================================================
// Host launcher
// =====================================================================
extern "C" void kernel_launch(void* const* d_in, const int* in_sizes, int n_in,
                              void* d_out, int out_size, void* d_ws, size_t ws_size,
                              hipStream_t stream) {
  (void)in_sizes; (void)n_in; (void)out_size; (void)ws_size;
  const float* x   = (const float*)d_in[0];
  const float* pos = (const float*)d_in[1];
  // d_in[2] = batch (sorted cloud ids) -- implied by layout, unused
  const float* W1  = (const float*)d_in[3];
  const float* b1  = (const float*)d_in[4];
  const float* W2  = (const float*)d_in[5];
  const float* b2  = (const float*)d_in[6];

  float* out      = (float*)d_out;                         // [B*S, HID]
  float* pos_out  = out + (size_t)BATCH * SSAMP * HIDD;    // [B*S, 3]
  int*   batch_out = (int*)(pos_out + (size_t)BATCH * SSAMP * 3);  // [B*S] int32
  int*   fpsidx   = (int*)d_ws;                            // [B*S] sample indices

  fps_kernel<<<BATCH, 1024, 0, stream>>>(pos, fpsidx, pos_out, batch_out);
  sa_kernel<<<(BATCH * SSAMP) / CPB, 256, 0, stream>>>(x, pos, W1, b1, W2, b2,
                                                       fpsidx, out);
}
